// Patchifier_558345748669
// MI455X (gfx1250) — matrix operand order repaired and checked by reading, not compiled, not run on hardware
//
#include <hip/hip_runtime.h>

// ---------------------------------------------------------------------------
// Patchifier for MI455X (gfx1250, wave32).
//  - 7x7/s4 convs (1->128ch) run as GEMM on the WMMA pipe:
//      D[128x16] = W[128x64(pad)] * im2col[64x16], two f16 16x16x32 K-steps,
//      f32 accumulation, fmap+imap share the im2col B operand.
//  - Everything else is bandwidth-bound streaming; frame (9.8MB) is
//    L2-resident (192MB L2), total traffic ~130MB -> ~10us floor @ 23.3TB/s.
// ---------------------------------------------------------------------------

typedef __attribute__((ext_vector_type(16))) _Float16 v16h;
typedef __attribute__((ext_vector_type(8)))  float    v8f;

#define IMGS 8
#define H    480
#define W    640
#define NCH  128
#define KSZ  7
#define KK   49          // KSZ*KSZ
#define KPAD 64          // K padded for two 16x16x32 WMMA steps
#define OH   120
#define OW   160
#define NP   192         // points per image
#define PS   7           // patch size
#define PGH  16
#define PGW  16
#define PH   30          // cell height (480/16)
#define PW   40          // cell width  (640/16)

// ---------------------------------------------------------------------------
// Kernel 0: pack w_f / w_c into 16-bit WMMA A fragments (contiguous k order, as B).
// packed[map][mtile(8)][kstep(2)][lane(32)][16 halves]
// lane<16 : M=lane,     halves = K(base+0..15)
// lane>=16: M=lane-16,  halves = K(base+16..31)
// K >= 49 is zero padding.
// ---------------------------------------------------------------------------
__global__ __launch_bounds__(256) void k_pack_weights(const float* __restrict__ w_f,
                                                      const float* __restrict__ w_c,
                                                      _Float16* __restrict__ packed) {
    int tid = blockIdx.x * 256 + threadIdx.x;          // 16384 total
    int h     =  tid        & 15;
    int lane  = (tid >> 4)  & 31;
    int kstep = (tid >> 9)  & 1;
    int mtile = (tid >> 10) & 7;
    int map   = (tid >> 13) & 1;
    int M = mtile * 16 + (lane & 15);
    int base = kstep * 32;
    int K;
    if (lane < 16) K = base + h;
    else           K = base + h + 16;
    const float* w = (map == 0) ? w_f : w_c;
    float v = (K < KK) ? w[M * KK + K] : 0.0f;
    packed[tid] = (_Float16)v;
}

// ---------------------------------------------------------------------------
// Kernel 1: encoder convs via WMMA.
// grid = (OW/16, OH, IMGS); block = 256 (8 waves).
// Each block: one row of 16 output pixels; wave w owns channels [16w,16w+16).
// ---------------------------------------------------------------------------
__global__ __launch_bounds__(256) void k_encoder(const float* __restrict__ frame,
                                                 const _Float16* __restrict__ wpack,
                                                 const float* __restrict__ b_f,
                                                 const float* __restrict__ b_c,
                                                 float* __restrict__ fmap,
                                                 float* __restrict__ imap) {
    const int tx  = blockIdx.x;        // 0..9  (x tile)
    const int oy  = blockIdx.y;        // 0..119
    const int img = blockIdx.z;        // 0..7
    const int tid = threadIdx.x;

    const float* fimg = frame + (size_t)img * H * W;

    // im2col B tile in LDS: [pixel n (16)][K (64, padded)], halves.
    __shared__ alignas(32) _Float16 b_lds[16][KPAD];

    // L2 prefetch of the next input row band (cheap hint, non-blocking).
    __builtin_prefetch(fimg + min(H - 1, 4 * oy + 8) * W + tx * 64, 0, 3);

    for (int e = tid; e < 16 * KPAD; e += 256) {       // exactly 4 iters/thread
        int n = e >> 6, k = e & 63;
        float v = 0.0f;
        if (k < KK) {
            int ky  = k / KSZ, kx = k % KSZ;
            int yin = 4 * oy - 3 + ky;
            int xin = 4 * (tx * 16 + n) - 3 + kx;
            if (yin >= 0 && yin < H && xin >= 0 && xin < W)
                v = fimg[yin * W + xin];
        }
        b_lds[n][k] = (_Float16)v;
    }
    __syncthreads();

    const int wave = tid >> 5;         // m-tile 0..7
    const int lane = tid & 31;

    // B fragments (contiguous k order, as A: lanes 0-15 hold K0..15 for
    // N=lane; lanes 16-31 hold K16..31 for N=lane-16 -> contiguous per lane).
    const int n  = lane & 15;
    const int kh = (lane >= 16) ? 16 : 0;
    v16h b0 = *(const v16h*)&b_lds[n][0  + kh];
    v16h b1 = *(const v16h*)&b_lds[n][32 + kh];

    // A fragments for both weight sets (L2-resident packed buffer).
    const v16h* wf = (const v16h*)wpack;               // [map][mtile][kstep][lane]
    v16h aF0 = wf[((0 * 8 + wave) * 2 + 0) * 32 + lane];
    v16h aF1 = wf[((0 * 8 + wave) * 2 + 1) * 32 + lane];
    v16h aC0 = wf[((1 * 8 + wave) * 2 + 0) * 32 + lane];
    v16h aC1 = wf[((1 * 8 + wave) * 2 + 1) * 32 + lane];

    v8f accF = {};
    v8f accC = {};
    accF = __builtin_amdgcn_wmma_f32_16x16x32_f16(false, aF0, false, b0, (short)0, accF, false, false);
    accF = __builtin_amdgcn_wmma_f32_16x16x32_f16(false, aF1, false, b1, (short)0, accF, false, false);
    accC = __builtin_amdgcn_wmma_f32_16x16x32_f16(false, aC0, false, b0, (short)0, accC, false, false);
    accC = __builtin_amdgcn_wmma_f32_16x16x32_f16(false, aC1, false, b1, (short)0, accC, false, false);

    // C/D layout: lanes 0-15 -> N=lane, M=r; lanes 16-31 -> N=lane-16, M=8+r.
    const int x     = tx * 16 + (lane & 15);
    const int mbase = wave * 16 + ((lane >= 16) ? 8 : 0);
#pragma unroll
    for (int r = 0; r < 8; ++r) {
        int ch = mbase + r;
        float vf = accF[r] + b_f[ch]; vf = vf > 0.0f ? vf : 0.0f;
        float vc = accC[r] + b_c[ch]; vc = vc > 0.0f ? vc : 0.0f;
        size_t o = (((size_t)img * NCH + ch) * OH + oy) * OW + x;
        fmap[o] = vf;
        imap[o] = vc;
    }
}

// ---------------------------------------------------------------------------
// Harris: gradients with reflect tail, 7x7 zero-padded box sums, response.
// ---------------------------------------------------------------------------
__device__ __forceinline__ float dxv(const float* f, int y, int x) {
    if (x >= W - 1) x = W - 3;                // reflect pad of the diff array
    return f[y * W + x + 1] - f[y * W + x];
}
__device__ __forceinline__ float dyv(const float* f, int y, int x) {
    if (y >= H - 1) y = H - 3;
    return f[(y + 1) * W + x] - f[y * W + x];
}

__global__ __launch_bounds__(256) void k_harris_h(const float* __restrict__ frame,
                                                  float* __restrict__ hx2,
                                                  float* __restrict__ hy2,
                                                  float* __restrict__ hxy) {
    size_t i = (size_t)blockIdx.x * 256 + threadIdx.x;   // IMGS*H*W exactly
    int x = i % W;
    int y = (i / W) % H;
    int img = i / ((size_t)W * H);
    const float* f = frame + (size_t)img * H * W;
    float sx = 0.0f, sy = 0.0f, sxy = 0.0f;
#pragma unroll
    for (int k = -3; k <= 3; ++k) {
        int xx = x + k;
        if (xx >= 0 && xx < W) {
            float a = dxv(f, y, xx);
            float b = dyv(f, y, xx);
            sx += a * a; sy += b * b; sxy += a * b;
        }
    }
    hx2[i] = sx; hy2[i] = sy; hxy[i] = sxy;
}

__global__ __launch_bounds__(256) void k_harris_v(const float* __restrict__ hx2,
                                                  const float* __restrict__ hy2,
                                                  const float* __restrict__ hxy,
                                                  float* __restrict__ g) {
    size_t i = (size_t)blockIdx.x * 256 + threadIdx.x;
    int x = i % W;
    int y = (i / W) % H;
    int img = i / ((size_t)W * H);
    size_t base = (size_t)img * H * W;
    float Ixx = 0.0f, Iyy = 0.0f, Ixy = 0.0f;
#pragma unroll
    for (int k = -3; k <= 3; ++k) {
        int yy = y + k;
        if (yy >= 0 && yy < H) {
            size_t j = base + (size_t)yy * W + x;
            Ixx += hx2[j]; Iyy += hy2[j]; Ixy += hxy[j];
        }
    }
    Ixx *= (1.0f / 49.0f); Iyy *= (1.0f / 49.0f); Ixy *= (1.0f / 49.0f);
    g[i] = (Ixx * Iyy - Ixy * Ixy) / (Ixx + Iyy + 1e-8f);
}

// ---------------------------------------------------------------------------
// Per-cell argmax (tie-break: lowest in-cell flat index, matching jnp.argmax).
// cells[img][cell] = {maxval, x, y}.  grid=(256 cells, IMGS)
// ---------------------------------------------------------------------------
__global__ __launch_bounds__(256) void k_cellmax(const float* __restrict__ g,
                                                 float* __restrict__ cells) {
    int cell = blockIdx.x, img = blockIdx.y;
    int gy = cell / PGW, gx = cell % PGW;
    const float* gi = g + (size_t)img * H * W;
    int tid = threadIdx.x;

    float bv = -3.402823466e+38f;
    int   bi = PH * PW;
    for (int e = tid; e < PH * PW; e += 256) {          // ascending -> stable
        int iy = e / PW, ix = e % PW;
        float v = gi[(size_t)(gy * PH + iy) * W + gx * PW + ix];
        if (v > bv) { bv = v; bi = e; }
    }
    __shared__ float sv[256];
    __shared__ int   si[256];
    sv[tid] = bv; si[tid] = bi;
    __syncthreads();
    for (int s = 128; s > 0; s >>= 1) {
        if (tid < s) {
            if (sv[tid + s] > sv[tid] ||
                (sv[tid + s] == sv[tid] && si[tid + s] < si[tid])) {
                sv[tid] = sv[tid + s]; si[tid] = si[tid + s];
            }
        }
        __syncthreads();
    }
    if (tid == 0) {
        int iy = si[0] / PW, ix = si[0] % PW;
        float* c = cells + ((size_t)img * 256 + cell) * 3;
        c[0] = sv[0];
        c[1] = (float)(gx * PW + ix);
        c[2] = (float)(gy * PH + iy);
    }
}

// ---------------------------------------------------------------------------
// Exact top-192 of 256 cells via rank counting (reproduces lax.top_k's
// stable descending order). grid=(IMGS), block=256.
// ---------------------------------------------------------------------------
__global__ __launch_bounds__(256) void k_topk(const float* __restrict__ cells,
                                              float* __restrict__ coords) {
    int img = blockIdx.x;
    int c   = threadIdx.x;
    __shared__ float v[256], xs[256], ys[256];
    const float* cc = cells + (size_t)img * 256 * 3;
    v[c] = cc[c * 3]; xs[c] = cc[c * 3 + 1]; ys[c] = cc[c * 3 + 2];
    __syncthreads();
    int rank = 0;
    for (int j = 0; j < 256; ++j)
        rank += (v[j] > v[c]) || (v[j] == v[c] && j < c);
    if (rank < NP) {
        float* o = coords + ((size_t)img * NP + rank) * 2;
        o[0] = xs[c];
        o[1] = ys[c];
    }
}

// ---------------------------------------------------------------------------
// Bilinear sampling (mask-then-clip, identical tap order to reference).
// ---------------------------------------------------------------------------
__device__ __forceinline__ float bilin(const float* __restrict__ plane,
                                       float xsf, float ysf) {
    float x0f = floorf(xsf), y0f = floorf(ysf);
    float wx = xsf - x0f, wy = ysf - y0f;
    int x0 = (int)x0f, y0 = (int)y0f;
    float out = 0.0f;
#pragma unroll
    for (int d = 0; d < 4; ++d) {
        int dxi = d & 1, dyi = d >> 1;
        int xi = x0 + dxi, yi = y0 + dyi;
        float wt = (dxi ? wx : 1.0f - wx) * (dyi ? wy : 1.0f - wy);
        float m  = (xi >= 0 && xi < OW && yi >= 0 && yi < OH) ? 1.0f : 0.0f;
        int xc = min(max(xi, 0), OW - 1);
        int yc = min(max(yi, 0), OH - 1);
        out += plane[yc * OW + xc] * (wt * m);
    }
    return out;
}

// grid=(NP, IMGS), block=256
__global__ __launch_bounds__(256) void k_patches(const float* __restrict__ coords,
                                                 const float* __restrict__ fmap,
                                                 const float* __restrict__ imap,
                                                 float* __restrict__ pf,
                                                 float* __restrict__ pc) {
    int p = blockIdx.x, img = blockIdx.y;
    const float* cpt = coords + ((size_t)img * NP + p) * 2;
    float cx = cpt[0] * 0.25f;     // coords / DS
    float cy = cpt[1] * 0.25f;

    for (int e = threadIdx.x; e < NCH * PS * PS; e += 256) {
        int c = e / (PS * PS), o = e % (PS * PS);
        float dy = (float)(o / PS) - 3.0f;    // o = i*7+j, i over dy, j over dx
        float dx = (float)(o % PS) - 3.0f;
        const float* plane = fmap + ((size_t)img * NCH + c) * OH * OW;
        pf[(((size_t)img * NP + p) * NCH + c) * (PS * PS) + o] =
            bilin(plane, cx + dx, cy + dy);
    }
    for (int c = threadIdx.x; c < NCH; c += 256) {
        const float* plane = imap + ((size_t)img * NCH + c) * OH * OW;
        pc[((size_t)img * NP + p) * NCH + c] = bilin(plane, cx, cy);
    }
}

// ---------------------------------------------------------------------------
// Host-side orchestration (all on `stream`, graph-capture safe).
// ---------------------------------------------------------------------------
extern "C" void kernel_launch(void* const* d_in, const int* in_sizes, int n_in,
                              void* d_out, int out_size, void* d_ws, size_t ws_size,
                              hipStream_t stream) {
    (void)in_sizes; (void)n_in; (void)out_size; (void)ws_size;
    const float* frame = (const float*)d_in[0];
    const float* w_f   = (const float*)d_in[1];
    const float* b_f   = (const float*)d_in[2];
    const float* w_c   = (const float*)d_in[3];
    const float* b_c   = (const float*)d_in[4];

    // d_out: coords | patches_f | patches_c | fmap  (all f32, flat)
    float* out    = (float*)d_out;
    float* coords = out;                                             // 8*192*2
    float* pf     = coords + (size_t)IMGS * NP * 2;                  // 8*192*128*49
    float* pc     = pf     + (size_t)IMGS * NP * NCH * PS * PS;      // 8*192*128
    float* fmap   = pc     + (size_t)IMGS * NP * NCH;                // 8*128*120*160

    // workspace layout (~113 MB): wpack | imap | hx2 | hy2 | hxy | g | cells
    float*    ws    = (float*)d_ws;
    _Float16* wpack = (_Float16*)ws;                                  // 16384 halves
    float*    imap  = ws   + 8192;
    float*    hx2   = imap + (size_t)IMGS * NCH * OH * OW;
    float*    hy2   = hx2  + (size_t)IMGS * H * W;
    float*    hxy   = hy2  + (size_t)IMGS * H * W;
    float*    g     = hxy  + (size_t)IMGS * H * W;
    float*    cells = g    + (size_t)IMGS * H * W;

    // 0) pack weights into WMMA A-fragment layout (16384 halves)
    k_pack_weights<<<64, 256, 0, stream>>>(w_f, w_c, wpack);

    // 1) both encoder convs via WMMA (fmap -> d_out, imap -> ws)
    k_encoder<<<dim3(OW / 16, OH, IMGS), 256, 0, stream>>>(
        frame, wpack, b_f, b_c, fmap, imap);

    // 2) Harris response (two fused passes)
    const int npix_blocks = (IMGS * H * W) / 256;      // 9600 exactly
    k_harris_h<<<npix_blocks, 256, 0, stream>>>(frame, hx2, hy2, hxy);
    k_harris_v<<<npix_blocks, 256, 0, stream>>>(hx2, hy2, hxy, g);

    // 3) per-cell argmax, then exact stable top-192 -> coords
    k_cellmax<<<dim3(PGH * PGW, IMGS), 256, 0, stream>>>(g, cells);
    k_topk<<<IMGS, 256, 0, stream>>>(cells, coords);

    // 4) bilinear patch extraction
    k_patches<<<dim3(NP, IMGS), 256, 0, stream>>>(coords, fmap, imap, pf, pc);
}
